// MultiHeadAttention_36593121362140
// MI455X (gfx1250) — matrix operand order repaired and checked by reading, not compiled, not run on hardware
//
#include <hip/hip_runtime.h>
#include <hip/hip_bf16.h>

typedef __attribute__((ext_vector_type(16))) _Float16 v16h;
typedef __attribute__((ext_vector_type(8)))  _Float16 v8h;
typedef __attribute__((ext_vector_type(8)))  float    v8f;

#define BDIM  2
#define SDIM  2048
#define DDIM  1024
#define HDIM  16
#define HD    64
#define N3    (3 * DDIM)

// ---------------------------------------------------------------------------
// Fragment loaders (CDNA5 WMMA 16x16x32 f16 layouts, wave32)
// A (16x32, MxK): lane m(0-15) -> row m, halves a[0..7]=K 0..7, a[8..15]=K 16..23
//                 lane m+16    -> row m, halves a[0..7]=K 8..15, a[8..15]=K 24..31
// B (32x16, KxN): lane n(0-15) -> col n, K 0..7,16..23 ; lane n+16 -> col n, K 8..15,24..31
// C (16x16):      VGPR r: lanes 0-15 -> row r, col=lane ; lanes 16-31 -> row r+8
// ---------------------------------------------------------------------------

__device__ __forceinline__ v16h load_a_frag_f16(const _Float16* A, int lda, int row0, int k0) {
    const int lane = threadIdx.x & 31;
    const int m = lane & 15;
    const int kb = k0 + ((lane & 16) ? 8 : 0);
    const _Float16* p = A + (size_t)(row0 + m) * lda + kb;
    v8h lo = *(const v8h*)p;
    v8h hi = *(const v8h*)(p + 16);
    v16h a;
#pragma unroll
    for (int i = 0; i < 8; ++i) { a[i] = lo[i]; a[8 + i] = hi[i]; }
    return a;
}

// B[k][n] where memory holds element [n][k] row-major (stride between n-rows):
// each lane reads two 8-f16 runs (K 8h..8h+7, 16+8h..). Used for all B fragments (transposed
// f16 weights, K rows for Q@K^T, V^T rows for P@V).
__device__ __forceinline__ v16h load_b_frag_rowgather_f16(const _Float16* P, int stride,
                                                          int nrow0, int k0) {
    const int lane = threadIdx.x & 31;
    const int n = lane & 15;
    const int kb = k0 + ((lane & 16) ? 8 : 0);
    const _Float16* p = P + (size_t)(nrow0 + n) * stride + kb;
    v8h lo = *(const v8h*)p;
    v8h hi = *(const v8h*)(p + 16);
    v16h b;
#pragma unroll
    for (int i = 0; i < 8; ++i) { b[i] = lo[i]; b[8 + i] = hi[i]; }
    return b;
}

__device__ __forceinline__ v8f wmma_f16(v16h a, v16h b, v8f c) {
    return __builtin_amdgcn_wmma_f32_16x16x32_f16(false, a, false, b, (short)0, c, false, false);
}

// ---------------------------------------------------------------------------
// One-shot precision/layout prep kernels (negligible vs GEMM cost, keeps the
// hot loops free of v_cvt and strided f32 loads).
// ---------------------------------------------------------------------------
__global__ void cvt_f32_to_f16(const float* __restrict__ X, _Float16* __restrict__ Xh, int n) {
    const int i = blockIdx.x * blockDim.x + threadIdx.x;
    if (i < n) Xh[i] = (_Float16)X[i];
}

// W[k][n] (K x N, f32) -> WT[n][k] (f16): coalesced reads, one-time scatter writes.
__global__ void cvt_transpose_f32_to_f16(const float* __restrict__ W, _Float16* __restrict__ WT,
                                         int K, int N) {
    const int i = blockIdx.x * blockDim.x + threadIdx.x;
    if (i < K * N) {
        const int k = i / N, n = i % N;
        WT[(size_t)n * K + k] = (_Float16)W[i];
    }
}

// ---------------------------------------------------------------------------
// Kernel 1: qkv = x @ w_qkv + b_qkv. One wave = 32x64 tile (8 accumulators,
// 8 WMMAs per K-step). Scatter to f16 Q/K [B,H,S,HD] and V^T [B,H,HD,S].
// ---------------------------------------------------------------------------
__global__ void mha_qkv_proj(const _Float16* __restrict__ xh, const _Float16* __restrict__ wT,
                             const float* __restrict__ b_qkv,
                             _Float16* __restrict__ q16, _Float16* __restrict__ k16,
                             _Float16* __restrict__ vT) {
    const int wave = (blockIdx.x * blockDim.x + threadIdx.x) >> 5;
    const int lane = threadIdx.x & 31;
    const int NT = N3 / 64;                 // 48 tiles along N
    const int mt = wave / NT;               // 0..127
    const int nt = wave % NT;
    const int row0 = mt * 32, col0 = nt * 64;

    v8f acc[2][4] = {};
    for (int k = 0; k < DDIM; k += 32) {
        const v16h a0 = load_a_frag_f16(xh, DDIM, row0,      k);
        const v16h a1 = load_a_frag_f16(xh, DDIM, row0 + 16, k);
#pragma unroll
        for (int j = 0; j < 4; ++j) {
            const v16h b = load_b_frag_rowgather_f16(wT, DDIM, col0 + j * 16, k);
            acc[0][j] = wmma_f16(a0, b, acc[0][j]);
            acc[1][j] = wmma_f16(a1, b, acc[1][j]);
        }
    }

    const int nlane = lane & 15, seg = (lane >> 4) & 1;
    const int sel = col0 >> 10;             // 0:q 1:k 2:v (64-tile never crosses)
    const int h   = (col0 >> 6) & 15;       // uniform across the 64-wide tile
#pragma unroll
    for (int j = 0; j < 4; ++j) {
        const int hd = j * 16 + nlane;
        const float bias = b_qkv[col0 + j * 16 + nlane];
#pragma unroll
        for (int i = 0; i < 2; ++i) {
#pragma unroll
            for (int r = 0; r < 8; ++r) {
                const int mg = row0 + i * 16 + r + seg * 8;
                const int bb = mg >> 11;            // / S
                const int ss = mg & (SDIM - 1);
                const _Float16 v = (_Float16)(acc[i][j][r] + bias);
                const size_t bh = (size_t)(bb * HDIM + h);
                if (sel == 0)      q16[(bh * SDIM + ss) * HD + hd] = v;
                else if (sel == 1) k16[(bh * SDIM + ss) * HD + hd] = v;
                else               vT [(bh * HD + hd) * SDIM + ss] = v;
            }
        }
    }
}

// ---------------------------------------------------------------------------
// Kernel 2: flash attention. One wave = one 16-query tile of one (b,h).
// ---------------------------------------------------------------------------
__global__ void mha_flash_attn(const _Float16* __restrict__ q16,
                               const _Float16* __restrict__ k16,
                               const _Float16* __restrict__ vT,
                               _Float16* __restrict__ aO) {
    __shared__ __align__(32) _Float16 plds[8][16 * 32];   // per-wave P staging (1KB each)

    const int wv   = threadIdx.x >> 5;
    const int lane = threadIdx.x & 31;
    const int wave = blockIdx.x * 8 + wv;
    const int QT = SDIM / 16;               // 128 query tiles per (b,h)
    const int qt = wave % QT;
    const int bh = wave / QT;               // 0..31
    const int bb = bh >> 4, hh = bh & 15;

    const _Float16* Q  = q16 + (size_t)bh * SDIM * HD;
    const _Float16* K  = k16 + (size_t)bh * SDIM * HD;
    const _Float16* VT = vT  + (size_t)bh * HD * SDIM;

    const v16h aq0 = load_a_frag_f16(Q, HD, qt * 16, 0);
    const v16h aq1 = load_a_frag_f16(Q, HD, qt * 16, 32);

    v8f o0 = {}, o1 = {}, o2 = {}, o3 = {};
    float m_i[8], l_i[8];
#pragma unroll
    for (int r = 0; r < 8; ++r) { m_i[r] = -1.0e30f; l_i[r] = 0.0f; }

    const int seg = (lane >> 4) & 1;
    const int n   = lane & 15;
    const float scale = 0.125f;             // 1/sqrt(64)

    for (int kb = 0; kb < SDIM; kb += 32) {
        // gfx1250 prefetch of the next key/value tiles (global_prefetch_b8)
        if (kb + 32 < SDIM) {
            __builtin_prefetch(K + (size_t)(kb + 32 + lane) * HD, 0, 0);
            __builtin_prefetch(VT + (size_t)(lane + 16 * seg) * SDIM + kb + 32, 0, 0);
        }

        // ---- S = scale * Q K^T  (16 x 32 keys) ----
        v8f s0 = {}, s1 = {};
        s0 = wmma_f16(aq0, load_b_frag_rowgather_f16(K, HD, kb, 0),       s0);
        s0 = wmma_f16(aq1, load_b_frag_rowgather_f16(K, HD, kb, 32),      s0);
        s1 = wmma_f16(aq0, load_b_frag_rowgather_f16(K, HD, kb + 16, 0),  s1);
        s1 = wmma_f16(aq1, load_b_frag_rowgather_f16(K, HD, kb + 16, 32), s1);

        // ---- online softmax ----
        float p0[8], p1[8], alpha[8];
#pragma unroll
        for (int r = 0; r < 8; ++r) {
            float v = fmaxf(s0[r], s1[r]) * scale;
#pragma unroll
            for (int off = 1; off < 16; off <<= 1)
                v = fmaxf(v, __shfl_xor(v, off, 16));
            const float mnew = fmaxf(m_i[r], v);
            alpha[r] = __expf(m_i[r] - mnew);
            p0[r] = __expf(s0[r] * scale - mnew);
            p1[r] = __expf(s1[r] * scale - mnew);
            float rs = p0[r] + p1[r];
#pragma unroll
            for (int off = 1; off < 16; off <<= 1)
                rs += __shfl_xor(rs, off, 16);
            l_i[r] = l_i[r] * alpha[r] + rs;
            m_i[r] = mnew;
        }
#pragma unroll
        for (int r = 0; r < 8; ++r) {
            o0[r] *= alpha[r]; o1[r] *= alpha[r];
            o2[r] *= alpha[r]; o3[r] *= alpha[r];
        }

        // ---- C-layout P -> A-layout via wave-private LDS ----
#pragma unroll
        for (int r = 0; r < 8; ++r) {
            const int row = r + seg * 8;
            plds[wv][row * 32 + n]      = (_Float16)p0[r];
            plds[wv][row * 32 + 16 + n] = (_Float16)p1[r];
        }
        asm volatile("s_wait_dscnt 0x0" ::: "memory");
        {
            const int kk = (lane & 16) ? 8 : 0;
            const _Float16* pp = &plds[wv][(lane & 15) * 32 + kk];
            v8h lo = *(const v8h*)pp;
            v8h hi = *(const v8h*)(pp + 16);
            v16h ap;
#pragma unroll
            for (int i = 0; i < 8; ++i) { ap[i] = lo[i]; ap[8 + i] = hi[i]; }

            // ---- O += P @ V  (V^T rows contiguous per lane) ----
            o0 = wmma_f16(ap, load_b_frag_rowgather_f16(VT, SDIM, 0,  kb), o0);
            o1 = wmma_f16(ap, load_b_frag_rowgather_f16(VT, SDIM, 16, kb), o1);
            o2 = wmma_f16(ap, load_b_frag_rowgather_f16(VT, SDIM, 32, kb), o2);
            o3 = wmma_f16(ap, load_b_frag_rowgather_f16(VT, SDIM, 48, kb), o3);
        }
    }

    // ---- normalize and store f16 into [B,S,D] for the output projection ----
#pragma unroll
    for (int r = 0; r < 8; ++r) {
        const float inv = 1.0f / l_i[r];
        const int srow = qt * 16 + r + seg * 8;
        _Float16* dst = aO + ((size_t)(bb * SDIM + srow)) * DDIM + hh * HD + n;
        dst[0]  = (_Float16)(o0[r] * inv);
        dst[16] = (_Float16)(o1[r] * inv);
        dst[32] = (_Float16)(o2[r] * inv);
        dst[48] = (_Float16)(o3[r] * inv);
    }
}

// ---------------------------------------------------------------------------
// Kernel 3: out = attnOut @ w_out + b_out. One wave = 32x64 tile.
// ---------------------------------------------------------------------------
__global__ void mha_out_proj(const _Float16* __restrict__ aO, const _Float16* __restrict__ wT,
                             const float* __restrict__ b_out, float* __restrict__ out) {
    const int wave = (blockIdx.x * blockDim.x + threadIdx.x) >> 5;
    const int lane = threadIdx.x & 31;
    const int NT = DDIM / 64;               // 16
    const int mt = wave / NT;               // 0..127
    const int nt = wave % NT;
    const int row0 = mt * 32, col0 = nt * 64;

    v8f acc[2][4] = {};
    for (int k = 0; k < DDIM; k += 32) {
        const v16h a0 = load_a_frag_f16(aO, DDIM, row0,      k);
        const v16h a1 = load_a_frag_f16(aO, DDIM, row0 + 16, k);
#pragma unroll
        for (int j = 0; j < 4; ++j) {
            const v16h b = load_b_frag_rowgather_f16(wT, DDIM, col0 + j * 16, k);
            acc[0][j] = wmma_f16(a0, b, acc[0][j]);
            acc[1][j] = wmma_f16(a1, b, acc[1][j]);
        }
    }

    const int nlane = lane & 15, seg = (lane >> 4) & 1;
#pragma unroll
    for (int j = 0; j < 4; ++j) {
        const int ncol = col0 + j * 16 + nlane;
        const float bias = b_out[ncol];
#pragma unroll
        for (int i = 0; i < 2; ++i) {
#pragma unroll
            for (int r = 0; r < 8; ++r) {
                const int mg = row0 + i * 16 + r + seg * 8;
                out[(size_t)mg * DDIM + ncol] = acc[i][j][r] + bias;
            }
        }
    }
}

// ---------------------------------------------------------------------------
extern "C" void kernel_launch(void* const* d_in, const int* in_sizes, int n_in,
                              void* d_out, int out_size, void* d_ws, size_t ws_size,
                              hipStream_t stream) {
    (void)in_sizes; (void)n_in; (void)out_size; (void)ws_size;
    const float* x     = (const float*)d_in[0];
    const float* w_qkv = (const float*)d_in[1];
    const float* b_qkv = (const float*)d_in[2];
    const float* w_out = (const float*)d_in[3];
    const float* b_out = (const float*)d_in[4];
    float* out = (float*)d_out;

    const size_t elems = (size_t)BDIM * HDIM * SDIM * HD;   // 4M halves per tensor
    _Float16* q16   = (_Float16*)d_ws;
    _Float16* k16   = q16 + elems;
    _Float16* vT    = k16 + elems;
    _Float16* aO    = vT  + elems;
    _Float16* xh    = aO  + elems;
    _Float16* wqkvT = xh  + elems;                           // D*3D = 3M halves
    _Float16* woutT = wqkvT + (size_t)DDIM * N3;             // D*D  = 1M halves

    const int xcount = BDIM * SDIM * DDIM;                   // 4M
    cvt_f32_to_f16<<<xcount / 256, 256, 0, stream>>>(x, xh, xcount);
    cvt_transpose_f32_to_f16<<<(DDIM * N3) / 256, 256, 0, stream>>>(w_qkv, wqkvT, DDIM, N3);
    cvt_transpose_f32_to_f16<<<(DDIM * DDIM) / 256, 256, 0, stream>>>(w_out, woutT, DDIM, DDIM);

    // 1) QKV projection: (4096/32)*(3072/64) = 6144 wave-tiles, 8 waves/block
    mha_qkv_proj<<<6144 / 8, 256, 0, stream>>>(xh, wqkvT, b_qkv, q16, k16, vT);
    // 2) Flash attention: 2*16*(2048/16) = 4096 wave-tiles
    mha_flash_attn<<<4096 / 8, 256, 0, stream>>>(q16, k16, vT, aO);
    // 3) Output projection: (4096/32)*(1024/64) = 2048 wave-tiles
    mha_out_proj<<<2048 / 8, 256, 0, stream>>>(aO, woutT, b_out, out);
}
